// CausalSelectiveSelfAttention_41772851921460
// MI455X (gfx1250) — matrix operand order repaired and checked by reading, not compiled, not run on hardware
//
#include <hip/hip_runtime.h>

// CausalSelectiveSelfAttention for MI455X (gfx1250, wave32, WMMA + async-LDS).
// Pipeline (all on `stream`, ws-backed):
//   1) cast x -> f16; transpose+cast W_attn, W_proj -> Bt layout [N][K] f16
//   2) WMMA GEMM (async-LDS B panel): qkv = x @ W_attn + b_attn   [B*T, 2304]
//   3) build vT f16 [B,H,D,T]
//   4) WMMA GEMM: att0 = q0 @ k0^T (raw dots, f32)                [B, T, T]
//   5) ff_cumsum: FF[t][s] = cumsum_t relu(mask(att0*0.125)) in-place
//   6) flash-style attention per (b,h,16-row tile): QK^T WMMA, -FF, online
//      softmax, P@V WMMA (P relayout via per-wave LDS tile) -> y f16
//   7) WMMA GEMM: out = y @ W_proj + b_proj (f32)

typedef __attribute__((ext_vector_type(16))) _Float16 v16h;
typedef __attribute__((ext_vector_type(8)))  _Float16 v8h;
typedef __attribute__((ext_vector_type(8)))  float    v8f;

#define BB 2
#define TT 2048
#define CC 768
#define HH 12
#define DD 64
#define C3 2304
#define NEG_BIG (-1.0e30f)

union V8F { v8f v; float f[8]; };

// A-fragment (16x32 f16, M x K): lane L holds row M = L%16.
// halves 0..7  -> K = 8*(L/16) + i        (16 contiguous bytes)
// halves 8..15 -> K = 16 + 8*(L/16) + i-8 (16 contiguous bytes)
__device__ __forceinline__ v16h frag_a16(const _Float16* p0, int ld, int lane) {
  int r = lane & 15, g = lane >> 4;
  const _Float16* p = p0 + (size_t)r * ld + 8 * g;
  union { v16h v; v8h h[2]; } u;
  u.h[0] = *(const v8h*)(p);
  u.h[1] = *(const v8h*)(p + 16);
  return u.v;
}

// B-fragment (32x16 f16, K x N) sourced from Bt row-major [N][K]:
// lane L holds column N = L%16; halves 0..7 -> K = 8*(L/16) + i,
// halves 8..15 -> K = 16 + 8*(L/16) + i-8
__device__ __forceinline__ v16h frag_b16(const _Float16* p0, int ld, int lane) {
  int n = lane & 15, g = lane >> 4;
  const _Float16* p = p0 + (size_t)n * ld + 8 * g;
  union { v16h v; v8h h[2]; } u;
  u.h[0] = *(const v8h*)(p);
  u.h[1] = *(const v8h*)(p + 16);
  return u.v;
}

__device__ __forceinline__ v8f wmma_f16(v16h a, v16h b, v8f c) {
  return __builtin_amdgcn_wmma_f32_16x16x32_f16(false, a, false, b, (short)0, c,
                                                false, false);
}

// CDNA5 async copy: LDS[vdst_lane] = MEM[vaddr_lane], 16B/lane, ASYNCcnt.
__device__ __forceinline__ void async_g2l_b128(unsigned lds_byte_addr,
                                               const void* gaddr) {
  asm volatile("global_load_async_to_lds_b128 %0, %1, off"
               :: "v"(lds_byte_addr), "v"((unsigned long long)(uintptr_t)gaddr)
               : "memory");
}
__device__ __forceinline__ void wait_asynccnt0() {
  asm volatile("s_wait_asynccnt 0x0" ::: "memory");
}

// ---------------- elementwise prep kernels ----------------

__global__ void cast_f16(const float* __restrict__ in, _Float16* __restrict__ out,
                         int n) {
  int i = blockIdx.x * blockDim.x + threadIdx.x;
  if (i < n) out[i] = (_Float16)in[i];
}

__global__ void transpose_cast_f16(const float* __restrict__ in,
                                   _Float16* __restrict__ out, int rows, int cols) {
  int i = blockIdx.x * blockDim.x + threadIdx.x;
  if (i >= rows * cols) return;
  int r = i / cols, c = i - r * cols;
  out[(size_t)c * rows + r] = (_Float16)in[i];
}

__global__ void build_vT(const _Float16* __restrict__ qkvh,
                         _Float16* __restrict__ vT) {
  int i = blockIdx.x * blockDim.x + threadIdx.x;
  if (i >= BB * HH * DD * TT) return;
  int t = i % TT;
  int d = (i / TT) % DD;
  int h = (i / (TT * DD)) % HH;
  int b = i / (TT * DD * HH);
  vT[i] = qkvh[((size_t)b * TT + t) * C3 + 2 * CC + (size_t)h * DD + d];
}

// ---------------- WMMA GEMM with async-LDS B staging ------------------------
// One 256-thread block (8 waves) computes a 128x64 C panel. The shared 64x32
// B panel per K-step is staged into double-buffered LDS with
// global_load_async_to_lds_b128 (16B per thread), overlapped with WMMA.
// All four B fragments are preloaded before the 4 back-to-back WMMAs so the
// ds_loads clause together and the XDL ops issue without per-op DScnt stalls.
// C[m][n] = sum_k A[m][k] * Bt[n][k] (+bias). M%128==0, K%32==0 required.

__global__ __launch_bounds__(256) void gemm_block(
    const _Float16* __restrict__ A, int lda,
    const _Float16* __restrict__ Bt, int ldb,
    const float* __restrict__ bias,
    float* __restrict__ Cf, _Float16* __restrict__ Ch, int ldc,
    int Ngroups, int K) {
  __shared__ __align__(64) _Float16 bufB[2][64 * 32];

  const int tid = threadIdx.x;
  const int wave = tid >> 5, lane = tid & 31;
  const int mb = blockIdx.x / Ngroups;          // 128-row panel index
  const int ng = blockIdx.x - mb * Ngroups;     // 64-col panel index

  const _Float16* a0 = A + (size_t)(mb * 128 + wave * 16) * lda;
  const _Float16* b0 = Bt + (size_t)(ng * 64) * ldb;

  // staging map: thread -> (row 0..63, 8-half chunk 0..3) of the B panel
  const int srow = tid >> 2;
  const int schk = (tid & 3) * 8;
  const _Float16* gB = b0 + (size_t)srow * ldb + schk;
  const unsigned ldsRowOff = (unsigned)(srow * 32 + schk) * 2u;
  const unsigned ldsBase0 = (unsigned)(uintptr_t)&bufB[0][0];
  const unsigned ldsBase1 = (unsigned)(uintptr_t)&bufB[1][0];

  V8F acc[4];
  for (int j = 0; j < 4; ++j)
    for (int e = 0; e < 8; ++e) acc[j].f[e] = 0.f;

  // prologue: stage k=0 into buffer 0
  async_g2l_b128(ldsBase0 + ldsRowOff, gB);

  for (int k = 0, it = 0; k < K; k += 32, ++it) {
    v16h a = frag_a16(a0 + k, lda, lane);       // issue A loads early
    if (k + 128 < K) __builtin_prefetch(a0 + k + 128, 0, 1);
    wait_asynccnt0();                           // my stage-k copies done
    __syncthreads();                            // everyone's copies done
    if (k + 32 < K)                             // overlap next stage copy
      async_g2l_b128(((it & 1) ? ldsBase0 : ldsBase1) + ldsRowOff, gB + k + 32);
    const _Float16* bb = &bufB[it & 1][0];
    v16h bfr[4];
#pragma unroll
    for (int j = 0; j < 4; ++j)                 // clause all 8 ds_loads
      bfr[j] = frag_b16(bb + j * 16 * 32, 32, lane);
#pragma unroll
    for (int j = 0; j < 4; ++j)                 // back-to-back XDL issue
      acc[j].v = wmma_f16(a, bfr[j], acc[j].v);
  }

  const int g = lane >> 4, cl = lane & 15;
  for (int j = 0; j < 4; ++j) {
    int col = ng * 64 + j * 16 + cl;
    float bv = bias ? bias[col] : 0.f;
    for (int e = 0; e < 8; ++e) {
      int row = mb * 128 + wave * 16 + e + 8 * g;
      float val = acc[j].f[e] + bv;
      if (Ch) Ch[(size_t)row * ldc + col] = (_Float16)val;
      else    Cf[(size_t)row * ldc + col] = val;
    }
  }
}

// ---------------- FF = cumsum over query rows of masked relu scores ---------

__global__ void ff_cumsum(float* __restrict__ att0) {
  int idx = blockIdx.x * blockDim.x + threadIdx.x;
  if (idx >= BB * TT) return;
  int b = idx / TT, s = idx - b * TT;
  float* p = att0 + (size_t)b * TT * TT + s;
  float acc = 0.f;
  for (int t = 0; t < TT; ++t) {
    float v = p[(size_t)t * TT] * 0.125f;
    float S = ((s < t) && (s != 0)) ? fmaxf(v, 0.f) : 0.f;  // causal,!bos,!self
    acc += S;
    p[(size_t)t * TT] = acc;
  }
}

// ---------------- flash-style attention with FF subtraction -----------------

__global__ __launch_bounds__(32) void attn_fwd(const _Float16* __restrict__ qkvh,
                                               const _Float16* __restrict__ vT,
                                               const float* __restrict__ FF,
                                               _Float16* __restrict__ yh) {
  const int lane = threadIdx.x;
  const int g = lane >> 4, cl = lane & 15;
  int id = blockIdx.x;
  int tb = id & 127;              // TT/16 == 128
  int h  = (id >> 7) % HH;
  int b  = id / (128 * HH);
  int row0 = tb * 16;

  const _Float16* qb = qkvh + (size_t)b * TT * C3 + (size_t)h * DD;
  const _Float16* kb = qkvh + (size_t)b * TT * C3 + CC + (size_t)h * DD;
  const _Float16* vb = vT + (size_t)(b * HH + h) * DD * TT;
  const float*   ffb = FF + (size_t)b * TT * TT;

  __shared__ __align__(32) _Float16 pT[16 * 32];

  float m[8], l[8];
  V8F o[4];
  for (int e = 0; e < 8; ++e) { m[e] = -3.0e30f; l[e] = 0.f; }
  for (int j = 0; j < 4; ++j)
    for (int e = 0; e < 8; ++e) o[j].f[e] = 0.f;

  v16h qa0 = frag_a16(qb + (size_t)row0 * C3 + 0,  C3, lane);
  v16h qa1 = frag_a16(qb + (size_t)row0 * C3 + 32, C3, lane);

  const int sEnd = row0 + 16;
  for (int s0 = 0; s0 < sEnd; s0 += 32) {
    // preload all 4 K fragments, then 4 back-to-back score WMMAs
    v16h kf[2][2];
#pragma unroll
    for (int hf = 0; hf < 2; ++hf) {
      kf[hf][0] = frag_b16(kb + (size_t)(s0 + 16 * hf) * C3 + 0,  C3, lane);
      kf[hf][1] = frag_b16(kb + (size_t)(s0 + 16 * hf) * C3 + 32, C3, lane);
    }
    V8F sc[2];
#pragma unroll
    for (int hf = 0; hf < 2; ++hf) {
      V8F t;
      for (int e = 0; e < 8; ++e) t.f[e] = 0.f;
      t.v = wmma_f16(qa0, kf[hf][0], t.v);
      t.v = wmma_f16(qa1, kf[hf][1], t.v);
      sc[hf] = t;
    }
    // scale, causal mask, subtract FF
    float sv[2][8];
#pragma unroll
    for (int hf = 0; hf < 2; ++hf)
      for (int e = 0; e < 8; ++e) {
        int r = e + 8 * g;
        int s = s0 + cl + 16 * hf;
        float v = sc[hf].f[e] * 0.125f;
        if (s > row0 + r) v = NEG_BIG;
        v -= ffb[(size_t)(row0 + r) * TT + s];
        sv[hf][e] = v;
      }
    // online softmax: each half-group of 16 lanes holds one full score row
    float mnew[8], scl[8];
    for (int e = 0; e < 8; ++e) {
      float v = fmaxf(sv[0][e], sv[1][e]);
      for (int off = 1; off < 16; off <<= 1)
        v = fmaxf(v, __shfl_xor(v, off, 32));
      mnew[e] = fmaxf(m[e], v);
      scl[e] = __expf(m[e] - mnew[e]);
    }
    for (int e = 0; e < 8; ++e) {
      int r = e + 8 * g;
      float p0 = __expf(sv[0][e] - mnew[e]);
      float p1 = __expf(sv[1][e] - mnew[e]);
      pT[r * 32 + cl]      = (_Float16)p0;
      pT[r * 32 + cl + 16] = (_Float16)p1;
      float rs = p0 + p1;
      for (int off = 1; off < 16; off <<= 1) rs += __shfl_xor(rs, off, 32);
      l[e] = l[e] * scl[e] + rs;
      m[e] = mnew[e];
    }
    for (int j = 0; j < 4; ++j)
      for (int e = 0; e < 8; ++e) o[j].f[e] *= scl[e];
    __syncthreads();  // single-wave WG: NOP barrier; LDS is in-order anyway
    v16h pa = frag_a16(pT, 32, lane);
    // preload all 4 V fragments, then 4 back-to-back PV WMMAs
    v16h vf[4];
#pragma unroll
    for (int j = 0; j < 4; ++j)
      vf[j] = frag_b16(vb + (size_t)(j * 16) * TT + s0, TT, lane);
#pragma unroll
    for (int j = 0; j < 4; ++j)
      o[j].v = wmma_f16(pa, vf[j], o[j].v);
    __syncthreads();
  }
  for (int j = 0; j < 4; ++j)
    for (int e = 0; e < 8; ++e) {
      int r = e + 8 * g;
      float val = o[j].f[e] / l[e];
      yh[((size_t)b * TT + row0 + r) * CC + (size_t)h * DD + j * 16 + cl] =
          (_Float16)val;
    }
}

// ---------------- launcher ----------------

extern "C" void kernel_launch(void* const* d_in, const int* in_sizes, int n_in,
                              void* d_out, int out_size, void* d_ws, size_t ws_size,
                              hipStream_t stream) {
  const float* x      = (const float*)d_in[0];
  const float* W_attn = (const float*)d_in[1];
  const float* b_attn = (const float*)d_in[2];
  const float* W_proj = (const float*)d_in[3];
  const float* b_proj = (const float*)d_in[4];
  float* out = (float*)d_out;

  char* w = (char*)d_ws;
  size_t off = 0;
  auto wsalloc = [&](size_t bytes) {
    char* p = w + off;
    off += (bytes + 255) & ~(size_t)255;
    return p;
  };
  _Float16* xh   = (_Float16*)wsalloc((size_t)BB * TT * CC * 2);
  _Float16* WaT  = (_Float16*)wsalloc((size_t)C3 * CC * 2);
  _Float16* WpT  = (_Float16*)wsalloc((size_t)CC * CC * 2);
  _Float16* qkvh = (_Float16*)wsalloc((size_t)BB * TT * C3 * 2);
  _Float16* vT   = (_Float16*)wsalloc((size_t)BB * HH * DD * TT * 2);
  float*    ffb  = (float*)wsalloc((size_t)BB * TT * TT * 4);
  _Float16* yh   = (_Float16*)wsalloc((size_t)BB * TT * CC * 2);
  (void)ws_size; (void)in_sizes; (void)n_in; (void)out_size;

  int n1 = BB * TT * CC;
  cast_f16<<<(n1 + 255) / 256, 256, 0, stream>>>(x, xh, n1);
  int n2 = CC * C3;
  transpose_cast_f16<<<(n2 + 255) / 256, 256, 0, stream>>>(W_attn, WaT, CC, C3);
  int n3 = CC * CC;
  transpose_cast_f16<<<(n3 + 255) / 256, 256, 0, stream>>>(W_proj, WpT, CC, CC);

  {  // qkv = x @ W_attn + b_attn  (f16 out)
    int Ngroups = C3 / 64;
    int blocks = (BB * TT / 128) * Ngroups;     // 32 * 36
    gemm_block<<<blocks, 256, 0, stream>>>(
        xh, CC, WaT, CC, b_attn, nullptr, qkvh, C3, Ngroups, CC);
  }
  int nv = BB * HH * DD * TT;
  build_vT<<<(nv + 255) / 256, 256, 0, stream>>>(qkvh, vT);

  for (int b = 0; b < BB; ++b) {  // att0 = q0 @ k0^T (raw dots, f32)
    const _Float16* A  = qkvh + (size_t)b * TT * C3;        // q head 0
    const _Float16* Bt = qkvh + (size_t)b * TT * C3 + CC;   // k head 0
    float* Cc = ffb + (size_t)b * TT * TT;
    int Ngroups = TT / 64;
    int blocks = (TT / 128) * Ngroups;          // 16 * 32
    gemm_block<<<blocks, 256, 0, stream>>>(
        A, C3, Bt, C3, nullptr, Cc, nullptr, TT, Ngroups, DD);
  }
  ff_cumsum<<<(BB * TT + 255) / 256, 256, 0, stream>>>(ffb);

  attn_fwd<<<BB * HH * (TT / 16), 32, 0, stream>>>(qkvh, vT, ffb, yh);

  {  // out = y @ W_proj + b_proj (f32)
    int Ngroups = CC / 64;
    int blocks = (BB * TT / 128) * Ngroups;     // 32 * 12
    gemm_block<<<blocks, 256, 0, stream>>>(
        yh, CC, WpT, CC, b_proj, out, nullptr, CC, Ngroups, CC);
  }
}
